// HydraScaleLM_8572754723197
// MI455X (gfx1250) — compile-verified, matrix operand order checked by reading, not run on hardware
//
#include <hip/hip_runtime.h>
#include <math.h>

// ---------------- model constants ----------------
#define EMBED   512
#define DEPTH   6
#define NSTATE  16
#define DCONV   4
#define DTR     32
#define PDIM    (DTR + 2*NSTATE)   // 64
#define VOCAB   50000
#define BATCH   2
#define SEQL    1024
#define BL      (BATCH*SEQL)       // 2048
#define MLPD    (4*EMBED)          // 2048

// ---------------- WMMA types ----------------
typedef __attribute__((ext_vector_type(16))) __bf16 v16bf;
typedef __attribute__((ext_vector_type(8)))  __bf16 v8bf;
typedef __attribute__((ext_vector_type(8)))  float  v8f;

// ---------------- activations ----------------
__device__ __forceinline__ float act_apply(float x, int act) {
    switch (act) {
        case 1: return 0.5f * x * (1.0f + erff(x * 0.70710678118654752f)); // exact GELU
        case 2: return x / (1.0f + __expf(-x));                           // SiLU
        case 3: return (x > 20.0f) ? x : log1pf(__expf(x));               // softplus
        default: return x;
    }
}

// ================= bf16 WMMA GEMM =================
// Out[M,N] = act( A[M,K](row stride lda) * W[K,N](row-major) + bias ) (+ Out if addC)
// Call-site invariants: M multiple of 128, K multiple of 32; only N may be ragged.
// Block tile 128x64, 8 waves: 4 in M (32 rows each) x 2 in N (32 cols each).
// Each wave: 2 A fragments x 2 B fragments -> 4 WMMAs per K-step.
#define TILE_M 128
#define TILE_N 64
#define TILE_K 32

__global__ __launch_bounds__(256) void gemm_bf16_wmma(
    const float* __restrict__ A, int lda,
    const float* __restrict__ W,
    const float* __restrict__ bias,
    float* __restrict__ Out, int ldo,
    int M, int N, int K, int act, int addC)
{
    // A tile row-major (k contiguous), W tile transposed (n-major, k contiguous):
    // each lane's fragment = two aligned 16-byte runs -> ds_load_b128 pairs.
    __shared__ __bf16 As [TILE_M][TILE_K];   // 8 KB
    __shared__ __bf16 WsT[TILE_N][TILE_K];   // 4 KB

    const int tid  = threadIdx.x;
    const int lane = tid & 31;
    const int wid  = tid >> 5;      // 0..7
    const int wm   = wid & 3;       // 32-row band within tile
    const int wn   = wid >> 2;      // 32-col band within tile

    const int blockM = blockIdx.y * TILE_M;
    const int blockN = blockIdx.x * TILE_N;

    // staging coordinates
    const int a_row = tid >> 2;              // 0..63 (+64 on second pass)
    const int a_cg  = (tid & 3) * 8;         // 0,8,16,24
    const int w_n   = tid & 63;              // 0..63 column within tile
    const int w_kg  = (tid >> 6) * 8;        // 0,8,16,24
    const int w_c   = blockN + w_n;
    const int w_cc  = (w_c < N) ? w_c : (N - 1);   // clamped, branch-free
    const float w_mask = (w_c < N) ? 1.0f : 0.0f;

    v8f acc00 = {}, acc01 = {}, acc10 = {}, acc11 = {};

    const int khalf = (lane >= 16) ? 8 : 0;
    const int arow0 = wm * 32 + (lane & 15);   // A fragment rows, low half
    const int arow1 = arow0 + 16;              // A fragment rows, high half
    const int bcol  = wn * 32 + (lane & 15);

    for (int k0 = 0; k0 < K; k0 += TILE_K) {
        // ---- stage A tile (128x32): 2 passes of 2x global_load_b128 + 16B LDS store
#pragma unroll
        for (int p = 0; p < 2; ++p) {
            int r = a_row + p * 64;
            const float4* ap =
                (const float4*)&A[(size_t)(blockM + r) * lda + (k0 + a_cg)];
            float4 a0 = ap[0];
            float4 a1 = ap[1];
            v8bf apk;
            apk[0] = (__bf16)a0.x; apk[1] = (__bf16)a0.y;
            apk[2] = (__bf16)a0.z; apk[3] = (__bf16)a0.w;
            apk[4] = (__bf16)a1.x; apk[5] = (__bf16)a1.y;
            apk[6] = (__bf16)a1.z; apk[7] = (__bf16)a1.w;
            *(v8bf*)&As[r][a_cg] = apk;
        }

        // ---- stage W tile transposed: 8 lane-coalesced loads + one 16B LDS store
        v8bf wpk;
#pragma unroll
        for (int j = 0; j < 8; ++j) {
            float v = W[(size_t)(k0 + w_kg + j) * N + w_cc] * w_mask;
            wpk[j] = (__bf16)v;
        }
        *(v8bf*)&WsT[w_n][w_kg] = wpk;

        __syncthreads();

        // prefetch next weight tile (global_prefetch_b8)
        if (k0 + TILE_K < K)
            __builtin_prefetch(&W[(size_t)(k0 + TILE_K + w_kg) * N + w_cc], 0, 1);

        // ---- vector fragment gather (2x b128 per fragment) ----
        v8bf a0lo = *(const v8bf*)&As [arow0]    [khalf];
        v8bf a0hi = *(const v8bf*)&As [arow0]    [khalf + 16];
        v8bf a1lo = *(const v8bf*)&As [arow1]    [khalf];
        v8bf a1hi = *(const v8bf*)&As [arow1]    [khalf + 16];
        v8bf b0lo = *(const v8bf*)&WsT[bcol]     [khalf];
        v8bf b0hi = *(const v8bf*)&WsT[bcol]     [khalf + 16];
        v8bf b1lo = *(const v8bf*)&WsT[bcol + 16][khalf];
        v8bf b1hi = *(const v8bf*)&WsT[bcol + 16][khalf + 16];

        v16bf afA = __builtin_shufflevector(a0lo, a0hi,
                      0,1,2,3,4,5,6,7,8,9,10,11,12,13,14,15);
        v16bf afB = __builtin_shufflevector(a1lo, a1hi,
                      0,1,2,3,4,5,6,7,8,9,10,11,12,13,14,15);
        v16bf bf0 = __builtin_shufflevector(b0lo, b0hi,
                      0,1,2,3,4,5,6,7,8,9,10,11,12,13,14,15);
        v16bf bf1 = __builtin_shufflevector(b1lo, b1hi,
                      0,1,2,3,4,5,6,7,8,9,10,11,12,13,14,15);

        acc00 = __builtin_amdgcn_wmma_f32_16x16x32_bf16(
                    false, afA, false, bf0, (short)0, acc00, false, false);
        acc01 = __builtin_amdgcn_wmma_f32_16x16x32_bf16(
                    false, afA, false, bf1, (short)0, acc01, false, false);
        acc10 = __builtin_amdgcn_wmma_f32_16x16x32_bf16(
                    false, afB, false, bf0, (short)0, acc10, false, false);
        acc11 = __builtin_amdgcn_wmma_f32_16x16x32_bf16(
                    false, afB, false, bf1, (short)0, acc11, false, false);

        __syncthreads();
    }

    // ---- epilogue: VGPR v -> M = band + v + (lane>=16 ? 8 : 0), N = lane&15 ----
    const int rlo = blockM + wm * 32 + khalf;        // rows for acc0x
    const int rhi = rlo + 16;                        // rows for acc1x
    const int c0  = blockN + wn * 32 + (lane & 15);
    const int c1  = c0 + 16;
    const float bias0 = (bias && c0 < N) ? bias[c0] : 0.0f;
    const float bias1 = (bias && c1 < N) ? bias[c1] : 0.0f;
#pragma unroll
    for (int v = 0; v < 8; ++v) {
        if (c0 < N) {
            float r0 = act_apply(acc00[v] + bias0, act);
            float r2 = act_apply(acc10[v] + bias0, act);
            size_t o0 = (size_t)(rlo + v) * ldo + c0;
            size_t o2 = (size_t)(rhi + v) * ldo + c0;
            if (addC) { r0 += Out[o0]; r2 += Out[o2]; }
            Out[o0] = r0;
            Out[o2] = r2;
        }
        if (c1 < N) {
            float r1 = act_apply(acc01[v] + bias1, act);
            float r3 = act_apply(acc11[v] + bias1, act);
            size_t o1 = (size_t)(rlo + v) * ldo + c1;
            size_t o3 = (size_t)(rhi + v) * ldo + c1;
            if (addC) { r1 += Out[o1]; r3 += Out[o3]; }
            Out[o1] = r1;
            Out[o3] = r3;
        }
    }
}

// ================= LayerNorm (one 512-wide row per block) =================
__global__ __launch_bounds__(256) void ln_kernel(
    const float* __restrict__ x, const float* __restrict__ s,
    const float* __restrict__ b, float* __restrict__ out)
{
    __shared__ float red0[8], red1[8];
    const int row = blockIdx.x;
    const float* xr = x + (size_t)row * EMBED;
    const int tid = threadIdx.x;
    float v0 = xr[tid];
    float v1 = xr[tid + 256];
    float sum = v0 + v1;
    float sq  = v0 * v0 + v1 * v1;
#pragma unroll
    for (int o = 16; o > 0; o >>= 1) {
        sum += __shfl_down(sum, o, 32);
        sq  += __shfl_down(sq,  o, 32);
    }
    if ((tid & 31) == 0) { red0[tid >> 5] = sum; red1[tid >> 5] = sq; }
    __syncthreads();
    float ts = 0.f, tq = 0.f;
#pragma unroll
    for (int i = 0; i < 8; ++i) { ts += red0[i]; tq += red1[i]; }
    const float mean = ts * (1.0f / EMBED);
    const float var  = tq * (1.0f / EMBED) - mean * mean;
    const float rstd = rsqrtf(var + 1e-5f);
    float* orow = out + (size_t)row * EMBED;
    orow[tid]       = (v0 - mean) * rstd * s[tid]       + b[tid];
    orow[tid + 256] = (v1 - mean) * rstd * s[tid + 256] + b[tid + 256];
}

// ================= time embedding (sinusoid + tiny MLP) =================
__global__ __launch_bounds__(256) void time_embed_kernel(
    const int* __restrict__ t,
    const float* __restrict__ w1, const float* __restrict__ b1,
    const float* __restrict__ w2, const float* __restrict__ b2,
    float* __restrict__ te)
{
    __shared__ float feat[EMBED];
    __shared__ float hbuf[MLPD];
    const int b = blockIdx.x;
    const float tv = (float)t[b];
    const float cexp = -logf(10000.0f) / 255.0f;   // half-1 = 255
    for (int j = threadIdx.x; j < 256; j += 256) {
        float a = tv * __expf((float)j * cexp);
        feat[j]       = __sinf(a);
        feat[j + 256] = __cosf(a);
    }
    __syncthreads();
    for (int u = threadIdx.x; u < MLPD; u += 256) {
        float acc = b1[u];
        for (int j = 0; j < EMBED; ++j)
            acc += feat[j] * w1[(size_t)j * MLPD + u];
        hbuf[u] = acc / (1.0f + __expf(-acc));     // SiLU
    }
    __syncthreads();
    for (int d = threadIdx.x; d < EMBED; d += 256) {
        float acc = b2[d];
        for (int j = 0; j < MLPD; ++j)
            acc += hbuf[j] * w2[(size_t)j * EMBED + d];
        te[(size_t)b * EMBED + d] = acc;
    }
}

// ================= token embed + time embed =================
__global__ __launch_bounds__(256) void embed_kernel(
    const int* __restrict__ tokens, const float* __restrict__ tok_emb,
    const float* __restrict__ te, float* __restrict__ x)
{
    int idx = blockIdx.x * 256 + threadIdx.x;       // 0 .. BL*EMBED
    int d  = idx & (EMBED - 1);
    int bl = idx >> 9;                              // EMBED = 512 = 2^9
    int b  = bl / SEQL;
    x[idx] = tok_emb[(size_t)tokens[bl] * EMBED + d] + te[(size_t)b * EMBED + d];
}

// ================= depthwise causal conv(4) + SiLU =================
__global__ __launch_bounds__(256) void conv_silu_kernel(
    const float* __restrict__ xp, const float* __restrict__ cw,
    float* __restrict__ u)
{
    int idx = blockIdx.x * 256 + threadIdx.x;
    int d  = idx & (EMBED - 1);
    int bl = idx >> 9;
    int l  = bl & (SEQL - 1);
    int b  = bl >> 10;
    float acc = 0.0f;
#pragma unroll
    for (int k = 0; k < DCONV; ++k) {
        int ls = l + k - (DCONV - 1);
        if (ls >= 0)
            acc += xp[(size_t)(b * SEQL + ls) * EMBED + d] * cw[d * DCONV + k];
    }
    u[idx] = acc / (1.0f + __expf(-acc));
}

// ================= selective scan over L, fused +u*D and SiLU(z) gate ======
__global__ __launch_bounds__(256) void scan_kernel(
    const float* __restrict__ dt, const float* __restrict__ u,
    const float* __restrict__ z,  const float* __restrict__ p,
    const float* __restrict__ A_log, const float* __restrict__ Dp,
    float* __restrict__ y)
{
    int g = blockIdx.x * 256 + threadIdx.x;   // B*EMBED chains
    if (g >= BATCH * EMBED) return;
    int b = g / EMBED;
    int d = g % EMBED;

    float A[NSTATE], h[NSTATE];
#pragma unroll
    for (int n = 0; n < NSTATE; ++n) {
        A[n] = -__expf(A_log[(size_t)d * NSTATE + n]);
        h[n] = 0.0f;
    }
    const float Dd = Dp[d];

    for (int l = 0; l < SEQL; ++l) {
        size_t e = (size_t)(b * SEQL + l) * EMBED + d;
        float dtv = dt[e];
        float uv  = u[e];
        float zv  = z[e];
        const float* pr = p + (size_t)(b * SEQL + l) * PDIM;
        float yv = 0.0f;
#pragma unroll
        for (int n = 0; n < NSTATE; ++n) {
            float At = __expf(A[n] * dtv);
            float Bt = (At - 1.0f) / (A[n] + 1e-8f);
            h[n] = At * h[n] + Bt * pr[DTR + n] * uv;
            yv  += pr[DTR + NSTATE + n] * h[n];
        }
        yv += uv * Dd;
        y[e] = yv * (zv / (1.0f + __expf(-zv)));
    }
}

// ================= host orchestration =================
extern "C" void kernel_launch(void* const* d_in, const int* in_sizes, int n_in,
                              void* d_out, int out_size, void* d_ws, size_t ws_size,
                              hipStream_t stream) {
    const int*   tokens  = (const int*)  d_in[0];
    const int*   t       = (const int*)  d_in[1];
    const float* tok_emb = (const float*)d_in[2];
    const float* time_w1 = (const float*)d_in[3];
    const float* time_b1 = (const float*)d_in[4];
    const float* time_w2 = (const float*)d_in[5];
    const float* time_b2 = (const float*)d_in[6];
    const float* n1_s    = (const float*)d_in[7];
    const float* n1_b    = (const float*)d_in[8];
    const float* wx      = (const float*)d_in[9];
    const float* wz      = (const float*)d_in[10];
    const float* wp      = (const float*)d_in[11];
    const float* conv_w  = (const float*)d_in[12];
    const float* dt_w    = (const float*)d_in[13];
    const float* dt_b    = (const float*)d_in[14];
    const float* A_log   = (const float*)d_in[15];
    const float* Dparam  = (const float*)d_in[16];
    const float* wo      = (const float*)d_in[17];
    const float* n2_s    = (const float*)d_in[18];
    const float* n2_b    = (const float*)d_in[19];
    const float* mlp_w1  = (const float*)d_in[20];
    const float* mlp_b1  = (const float*)d_in[21];
    const float* mlp_w2  = (const float*)d_in[22];
    const float* mlp_b2  = (const float*)d_in[23];
    const float* no_s    = (const float*)d_in[24];
    const float* no_b    = (const float*)d_in[25];
    const float* head_w  = (const float*)d_in[26];
    const float* head_b  = (const float*)d_in[27];

    // workspace carve-up (floats)
    float* ws = (float*)d_ws;
    size_t off = 0;
    float* x   = ws + off; off += (size_t)BL * EMBED;
    float* h   = ws + off; off += (size_t)BL * EMBED;
    float* xp  = ws + off; off += (size_t)BL * EMBED;
    float* zb  = ws + off; off += (size_t)BL * EMBED;
    float* pb  = ws + off; off += (size_t)BL * PDIM;
    float* dtb = ws + off; off += (size_t)BL * EMBED;
    float* ub  = ws + off; off += (size_t)BL * EMBED;
    float* yb  = ws + off; off += (size_t)BL * EMBED;
    float* hid = ws + off; off += (size_t)BL * MLPD;
    float* te  = ws + off; off += (size_t)BATCH * EMBED;

    const dim3 blk(256);
    const dim3 gE(EMBED / TILE_N, BL / TILE_M);   // (8, 16)  N=512
    const dim3 gP(PDIM  / TILE_N, BL / TILE_M);   // (1, 16)  N=64
    const dim3 gH(MLPD  / TILE_N, BL / TILE_M);   // (32, 16) N=2048
    const dim3 gV((VOCAB + TILE_N - 1) / TILE_N, BL / TILE_M); // (782, 16)
    const int  nElem = BL * EMBED / 256;          // 4096 blocks

    time_embed_kernel<<<BATCH, blk, 0, stream>>>(t, time_w1, time_b1, time_w2, time_b2, te);
    embed_kernel<<<nElem, blk, 0, stream>>>(tokens, tok_emb, te, x);

    for (int i = 0; i < DEPTH; ++i) {
        const float* wx_i  = wx     + (size_t)i * EMBED * EMBED;
        const float* wz_i  = wz     + (size_t)i * EMBED * EMBED;
        const float* wp_i  = wp     + (size_t)i * EMBED * PDIM;
        const float* cw_i  = conv_w + (size_t)i * EMBED * DCONV;
        const float* dtw_i = dt_w   + (size_t)i * DTR * EMBED;
        const float* dtb_i = dt_b   + (size_t)i * EMBED;
        const float* Al_i  = A_log  + (size_t)i * EMBED * NSTATE;
        const float* Dp_i  = Dparam + (size_t)i * EMBED;
        const float* wo_i  = wo     + (size_t)i * EMBED * EMBED;
        const float* w1_i  = mlp_w1 + (size_t)i * EMBED * MLPD;
        const float* b1_i  = mlp_b1 + (size_t)i * MLPD;
        const float* w2_i  = mlp_w2 + (size_t)i * MLPD * EMBED;
        const float* b2_i  = mlp_b2 + (size_t)i * EMBED;

        // ---- SSM branch ----
        ln_kernel<<<BL, blk, 0, stream>>>(x, n1_s + i * EMBED, n1_b + i * EMBED, h);
        gemm_bf16_wmma<<<gE, blk, 0, stream>>>(h, EMBED, wx_i, nullptr, xp, EMBED,
                                               BL, EMBED, EMBED, 0, 0);
        gemm_bf16_wmma<<<gE, blk, 0, stream>>>(h, EMBED, wz_i, nullptr, zb, EMBED,
                                               BL, EMBED, EMBED, 0, 0);
        gemm_bf16_wmma<<<gP, blk, 0, stream>>>(h, EMBED, wp_i, nullptr, pb, PDIM,
                                               BL, PDIM, EMBED, 0, 0);
        // dt = softplus(dtu @ dt_w + dt_b), dtu = pb[:, :32]
        gemm_bf16_wmma<<<gE, blk, 0, stream>>>(pb, PDIM, dtw_i, dtb_i, dtb, EMBED,
                                               BL, EMBED, DTR, 3, 0);
        conv_silu_kernel<<<nElem, blk, 0, stream>>>(xp, cw_i, ub);
        scan_kernel<<<(BATCH * EMBED + 255) / 256, blk, 0, stream>>>(
            dtb, ub, zb, pb, Al_i, Dp_i, yb);
        // x += y @ wo
        gemm_bf16_wmma<<<gE, blk, 0, stream>>>(yb, EMBED, wo_i, nullptr, x, EMBED,
                                               BL, EMBED, EMBED, 0, 1);

        // ---- MLP branch ----
        ln_kernel<<<BL, blk, 0, stream>>>(x, n2_s + i * EMBED, n2_b + i * EMBED, h);
        gemm_bf16_wmma<<<gH, blk, 0, stream>>>(h, EMBED, w1_i, b1_i, hid, MLPD,
                                               BL, MLPD, EMBED, 1, 0);       // GELU
        gemm_bf16_wmma<<<gE, blk, 0, stream>>>(hid, MLPD, w2_i, b2_i, x, EMBED,
                                               BL, EMBED, MLPD, 0, 1);       // += residual
    }

    // ---- head ----
    ln_kernel<<<BL, blk, 0, stream>>>(x, no_s, no_b, h);
    gemm_bf16_wmma<<<gV, blk, 0, stream>>>(h, EMBED, head_w, head_b,
                                           (float*)d_out, VOCAB,
                                           BL, VOCAB, EMBED, 0, 0);
}
